// OptimizedECGEmotionTransformer_81063212744886
// MI455X (gfx1250) — matrix operand order repaired and checked by reading, not compiled, not run on hardware
//
#include <hip/hip_runtime.h>
#include <hip/hip_bf16.h>
#include <math.h>

// ---------------------------------------------------------------------------
// CDNA5 (gfx1250, wave32) types
// ---------------------------------------------------------------------------
typedef __attribute__((ext_vector_type(16))) __bf16 bf16x16;
typedef __attribute__((ext_vector_type(8)))  __bf16 bf16x8;
typedef __attribute__((ext_vector_type(8)))  float  f32x8;

union FragBF { bf16x16 v; bf16x8 h[2]; };

#define EPI_NONE     0
#define EPI_GELU     1
#define EPI_SIG      2
#define EPI_RELU     3
#define EPI_SOFTPLUS 4

// Async global->LDS staging (CDNA5): gate on builtin availability so the file
// compiles on toolchains without the async builtins (falls back to ds_store).
#if __has_builtin(__builtin_amdgcn_global_load_async_to_lds_b128) && \
    __has_builtin(__builtin_amdgcn_s_wait_asynccnt)
#define USE_ASYNC_LDS 1
typedef __attribute__((ext_vector_type(4))) int i32x4_t;
typedef __attribute__((address_space(1))) i32x4_t* as1_v4p;  // global int4*
typedef __attribute__((address_space(3))) i32x4_t* as3_v4p;  // LDS int4*
#else
#define USE_ASYNC_LDS 0
#endif

__device__ inline unsigned short f2bf(float f) {
  // round-to-nearest-even f32 -> bf16
  unsigned int u = __float_as_uint(f);
  u += 0x7FFFu + ((u >> 16) & 1u);
  return (unsigned short)(u >> 16);
}

__device__ inline float epi_f(float x, int e) {
  switch (e) {
    case EPI_GELU:     return 0.5f * x * (1.f + erff(x * 0.70710678118654752f));
    case EPI_SIG:      return 1.f / (1.f + __expf(-x));
    case EPI_RELU:     return fmaxf(x, 0.f);
    case EPI_SOFTPLUS: return (x > 20.f) ? x : log1pf(__expf(x));
    default:           return x;
  }
}

// 16-byte global -> LDS copy (async on CDNA5, sync fallback otherwise)
__device__ inline void copy16(unsigned short* lds, const unsigned short* g) {
#if USE_ASYNC_LDS
  __builtin_amdgcn_global_load_async_to_lds_b128((as1_v4p)g, (as3_v4p)lds, 0, 0);
#else
  *reinterpret_cast<uint4*>(lds) = *reinterpret_cast<const uint4*>(g);
#endif
}

// ---------------------------------------------------------------------------
// bf16 WMMA GEMM:  C[M,N] = epi( A[M,K] @ W[K,N] + bias[N] )
// A  : bf16 row-major  [M][K]
// Wt : bf16 TRANSPOSED weights, row-major [N][K] (pre-converted on device)
// Outputs: Cf (fp32, optional) and/or Cbf (bf16 shadow for chained GEMMs).
// Block tile 128x128, K-step 32, 256 threads = 8 wave32s (each wave 32x64).
// LDS double-buffered; staging via GLOBAL_LOAD_ASYNC_TO_LDS_B128 (ASYNCcnt).
// rowmap==1: A row r -> r + r/448 + 1 (skip-cls gather for pooling K/V).
// NOTE: tile staging reads up to 127 rows past M -> all bf16 buffers padded.
// ---------------------------------------------------------------------------
__global__ __launch_bounds__(256) void k_gemm(
    const unsigned short* __restrict__ A, const unsigned short* __restrict__ Wt,
    const float* __restrict__ bias, float* __restrict__ Cf,
    unsigned short* __restrict__ Cbf, int M, int N, int K, int epi, int rowmap)
{
  __shared__ unsigned short Al[2][128][32];
  __shared__ unsigned short Bl[2][128][32];

  const int tid  = threadIdx.x;
  const int mt   = blockIdx.y, nt = blockIdx.x;
  const int lane = tid & 31,  wid = tid >> 5;
  const int wm   = wid >> 1,  wn  = wid & 1;
  const int rbase = wm * 32,  cbase = wn * 64;
  const int half  = lane >> 4, l16 = lane & 15;

  f32x8 acc[2][4];
  const f32x8 vz = {0.f,0.f,0.f,0.f,0.f,0.f,0.f,0.f};
#pragma unroll
  for (int i = 0; i < 2; ++i)
#pragma unroll
    for (int j = 0; j < 4; ++j) acc[i][j] = vz;

  // stage one 128x32 A-tile and 128x32 B-tile (16B chunks, 4 copies/thread)
  auto stage = [&](int kt, int bufi) {
    const int k0 = kt * 32;
#pragma unroll
    for (int i = 0; i < 2; ++i) {
      int idx = tid + i * 256;               // 0..511 16B slots
      int row = idx >> 2, c = (idx & 3) * 8; // 8 bf16 per slot
      int arow = mt * 128 + row;
      int ar = rowmap ? (arow + arow / 448 + 1) : arow;
      copy16(&Al[bufi][row][c], A + (size_t)ar * K + k0 + c);
    }
#pragma unroll
    for (int i = 0; i < 2; ++i) {
      int idx = tid + i * 256;
      int row = idx >> 2, c = (idx & 3) * 8;
      copy16(&Bl[bufi][row][c], Wt + (size_t)(nt * 128 + row) * K + k0 + c);
    }
  };

  const int KT = K >> 5;
  stage(0, 0);
  for (int kt = 0; kt < KT; ++kt) {
    const int cur = kt & 1;
    if (kt + 1 < KT) {
      stage(kt + 1, cur ^ 1);
#if USE_ASYNC_LDS
      __builtin_amdgcn_s_wait_asynccnt(4);   // current tile's 4 copies done
#endif
    } else {
#if USE_ASYNC_LDS
      __builtin_amdgcn_s_wait_asynccnt(0);
#endif
    }
    __syncthreads();                          // all waves' tile data visible

    FragBF fa[2], fb[4];
#pragma unroll
    for (int im = 0; im < 2; ++im) {
      const unsigned short* p = &Al[cur][rbase + im * 16 + l16][0];
      fa[im].h[0] = *reinterpret_cast<const bf16x8*>(p + half * 8);
      fa[im].h[1] = *reinterpret_cast<const bf16x8*>(p + 16 + half * 8);
    }
#pragma unroll
    for (int jn = 0; jn < 4; ++jn) {
      const unsigned short* p = &Bl[cur][cbase + jn * 16 + l16][0];
      fb[jn].h[0] = *reinterpret_cast<const bf16x8*>(p + half * 8);
      fb[jn].h[1] = *reinterpret_cast<const bf16x8*>(p + 16 + half * 8);
    }
#pragma unroll
    for (int im = 0; im < 2; ++im)
#pragma unroll
      for (int jn = 0; jn < 4; ++jn)
        acc[im][jn] = __builtin_amdgcn_wmma_f32_16x16x32_bf16(
            false, fa[im].v, false, fb[jn].v, (short)0, acc[im][jn], false, false);

    __syncthreads();                          // safe to overwrite buf cur
  }

  // ---- epilogue (C/D layout: lane 0-15 N, VGPR v -> M = v + 8*(lane>=16)) ----
#pragma unroll
  for (int im = 0; im < 2; ++im) {
#pragma unroll
    for (int jn = 0; jn < 4; ++jn) {
      int col = nt * 128 + cbase + jn * 16 + l16;
      float bv = bias ? bias[col] : 0.f;
#pragma unroll
      for (int v = 0; v < 8; ++v) {
        int row = mt * 128 + rbase + im * 16 + v + half * 8;
        if (row < M) {
          float xv = epi_f(acc[im][jn][v] + bv, epi);
          if (Cf)  Cf[(size_t)row * N + col] = xv;
          if (Cbf) Cbf[(size_t)row * N + col] = f2bf(xv);
        }
      }
    }
  }
}

// ---------------------------------------------------------------------------
// One-time weight convert+transpose: Wt[n*K+k] = bf16(W[k*N+n]); blockIdx.y=mat
// ---------------------------------------------------------------------------
__global__ void k_w2bf_t(const float* __restrict__ W, unsigned short* __restrict__ Wt,
                         int K, int N)
{
  size_t per = (size_t)K * N;
  size_t idx = (size_t)blockIdx.x * 256 + threadIdx.x;
  if (idx >= per) return;
  int k = (int)(idx / N), n = (int)(idx % N);
  size_t mo = (size_t)blockIdx.y * per;
  Wt[mo + (size_t)n * K + k] = f2bf(W[mo + (size_t)k * N + n]);
}

// elementwise fp32 -> bf16
__global__ void k_f2bf(const float* __restrict__ src, unsigned short* __restrict__ dst,
                       long n)
{
  long i = (long)blockIdx.x * 256 + threadIdx.x;
  if (i < n) dst[i] = f2bf(src[i]);
}

// ---------------------------------------------------------------------------
// Scalar GEMM for tiny N (fp32): out[m*ostride+n] = epi(A@W + bias)
// ---------------------------------------------------------------------------
__global__ void k_small(const float* __restrict__ A, const float* __restrict__ W,
                        const float* __restrict__ bias, float* __restrict__ outp,
                        int M, int N, int K, int ostride, int epi)
{
  int id = blockIdx.x * 256 + threadIdx.x;
  if (id >= M * N) return;
  int m = id / N, n = id % N;
  float acc = bias ? bias[n] : 0.f;
  for (int k = 0; k < K; ++k)
    acc += A[(size_t)m * K + k] * W[(size_t)k * N + n];
  outp[(size_t)m * ostride + n] = epi_f(acc, epi);
}

// ---------------------------------------------------------------------------
// Block LN reduction helper (256 threads)
// ---------------------------------------------------------------------------
__device__ inline void red2(float* r1, float* r2, int tid, float s, float s2,
                            float inv_n, float& mean, float& rinv)
{
  r1[tid] = s; r2[tid] = s2; __syncthreads();
  for (int st = 128; st > 0; st >>= 1) {
    if (tid < st) { r1[tid] += r1[tid + st]; r2[tid] += r2[tid + st]; }
    __syncthreads();
  }
  mean = r1[0] * inv_n;
  float var = r2[0] * inv_n - mean * mean;
  rinv = rsqrtf(var + 1e-5f);
  __syncthreads();
}

// generic rowwise LN; fp32 and/or bf16 outputs (either may be null)
__global__ void k_layernorm(const float* __restrict__ src, float* __restrict__ dstf,
                            unsigned short* __restrict__ dstb,
                            const float* __restrict__ w, const float* __restrict__ b,
                            int cols)
{
  int r = blockIdx.x, tid = threadIdx.x;
  __shared__ float r1[256], r2[256];
  float s = 0.f, s2 = 0.f;
  for (int c = tid; c < cols; c += 256) {
    float v = src[(size_t)r * cols + c];
    s += v; s2 += v * v;
  }
  float mean, rinv;
  red2(r1, r2, tid, s, s2, 1.f / (float)cols, mean, rinv);
  for (int c = tid; c < cols; c += 256) {
    float v = (src[(size_t)r * cols + c] - mean) * rinv * w[c] + b[c];
    if (dstf) dstf[(size_t)r * cols + c] = v;
    if (dstb) dstb[(size_t)r * cols + c] = f2bf(v);
  }
}

// ---------------------------------------------------------------------------
// h = concat(cls, LN(tmp)+pos) ; rows = B*449 ; writes fp32 + bf16 shadow
// ---------------------------------------------------------------------------
__global__ void k_build_h(const float* __restrict__ tmp, const float* __restrict__ cls,
                          const float* __restrict__ pos, const float* __restrict__ w,
                          const float* __restrict__ bb, float* __restrict__ Hout,
                          unsigned short* __restrict__ Hbf)
{
  int r = blockIdx.x, tid = threadIdx.x;
  int bidx = r / 449, s = r % 449;
  if (s == 0) {
    for (int c = tid; c < 512; c += 256) {
      float v = cls[c];
      Hout[(size_t)r * 512 + c] = v;
      Hbf [(size_t)r * 512 + c] = f2bf(v);
    }
    return;
  }
  const float* xr = tmp + (size_t)(bidx * 448 + s - 1) * 512;
  __shared__ float r1[256], r2[256];
  float v0 = xr[tid], v1 = xr[tid + 256];
  float mean, rinv;
  red2(r1, r2, tid, v0 + v1, v0 * v0 + v1 * v1, 1.f / 512.f, mean, rinv);
  const float* pr = pos + (size_t)(s - 1) * 512;
  float o0 = (v0 - mean) * rinv * w[tid]       + bb[tid]       + pr[tid];
  float o1 = (v1 - mean) * rinv * w[tid + 256] + bb[tid + 256] + pr[tid + 256];
  Hout[(size_t)r * 512 + tid]       = o0;  Hbf[(size_t)r * 512 + tid]       = f2bf(o0);
  Hout[(size_t)r * 512 + tid + 256] = o1;  Hbf[(size_t)r * 512 + tid + 256] = f2bf(o1);
}

// ---------------------------------------------------------------------------
// attn mid: t = h + ga + 0.3*conv(h); a = LN(t,aln); Hout = LN(h + a, n1)
// ---------------------------------------------------------------------------
__global__ void k_attn_mid(const float* __restrict__ Hin, const float* __restrict__ ga,
                           const float* __restrict__ cw, const float* __restrict__ cb,
                           const float* __restrict__ aw, const float* __restrict__ ab,
                           const float* __restrict__ n1w, const float* __restrict__ n1b,
                           float* __restrict__ Hout, unsigned short* __restrict__ Hbf)
{
  int r = blockIdx.x, tid = threadIdx.x;
  int s = r % 449;
  __shared__ float r1[256], r2[256];
  float t0, t1, h0, h1;
  {
    size_t base = (size_t)r * 512 + tid;
    float hm = (s > 0)   ? Hin[base - 512] : 0.f;
    h0 = Hin[base];
    float hp = (s < 448) ? Hin[base + 512] : 0.f;
    float la = cb[tid] + hm * cw[tid*3] + h0 * cw[tid*3+1] + hp * cw[tid*3+2];
    t0 = h0 + ga[base] + 0.3f * la;
  }
  {
    int d = tid + 256;
    size_t base = (size_t)r * 512 + d;
    float hm = (s > 0)   ? Hin[base - 512] : 0.f;
    h1 = Hin[base];
    float hp = (s < 448) ? Hin[base + 512] : 0.f;
    float la = cb[d] + hm * cw[d*3] + h1 * cw[d*3+1] + hp * cw[d*3+2];
    t1 = h1 + ga[base] + 0.3f * la;
  }
  float mean, rinv;
  red2(r1, r2, tid, t0 + t1, t0 * t0 + t1 * t1, 1.f / 512.f, mean, rinv);
  float u0 = h0 + ((t0 - mean) * rinv * aw[tid]       + ab[tid]);
  float u1 = h1 + ((t1 - mean) * rinv * aw[tid + 256] + ab[tid + 256]);
  red2(r1, r2, tid, u0 + u1, u0 * u0 + u1 * u1, 1.f / 512.f, mean, rinv);
  float o0 = (u0 - mean) * rinv * n1w[tid]       + n1b[tid];
  float o1 = (u1 - mean) * rinv * n1w[tid + 256] + n1b[tid + 256];
  Hout[(size_t)r * 512 + tid]       = o0;  Hbf[(size_t)r * 512 + tid]       = f2bf(o0);
  Hout[(size_t)r * 512 + tid + 256] = o1;  Hbf[(size_t)r * 512 + tid + 256] = f2bf(o1);
}

// ---------------------------------------------------------------------------
// ffn end: Hout = LN(H2 + f3*gate, n2)  (fp32 + bf16 shadow)
// ---------------------------------------------------------------------------
__global__ void k_ffn_end(const float* __restrict__ H2, const float* __restrict__ f3,
                          const float* __restrict__ gate, const float* __restrict__ w,
                          const float* __restrict__ bb, float* __restrict__ Hout,
                          unsigned short* __restrict__ Hbf)
{
  int r = blockIdx.x, tid = threadIdx.x;
  __shared__ float r1[256], r2[256];
  size_t base = (size_t)r * 512;
  float u0 = H2[base + tid]       + f3[base + tid]       * gate[base + tid];
  float u1 = H2[base + tid + 256] + f3[base + tid + 256] * gate[base + tid + 256];
  float mean, rinv;
  red2(r1, r2, tid, u0 + u1, u0 * u0 + u1 * u1, 1.f / 512.f, mean, rinv);
  float o0 = (u0 - mean) * rinv * w[tid]       + bb[tid];
  float o1 = (u1 - mean) * rinv * w[tid + 256] + bb[tid + 256];
  Hout[base + tid]       = o0;  Hbf[base + tid]       = f2bf(o0);
  Hout[base + tid + 256] = o1;  Hbf[base + tid + 256] = f2bf(o1);
}

// ---------------------------------------------------------------------------
// Main attention (online softmax; one q-row/thread; K/V chunks in LDS);
// writes bf16 directly (consumed only by the W_o GEMM).
// grid: (2, H=8, B=64), block 256
// ---------------------------------------------------------------------------
__global__ __launch_bounds__(256) void k_attn(const float* __restrict__ Qp,
                                              const float* __restrict__ Kp,
                                              const float* __restrict__ Vp,
                                              unsigned short* __restrict__ Obf)
{
  int b = blockIdx.z, hh = blockIdx.y, qb = blockIdx.x;
  int tid = threadIdx.x;
  int q = qb * 256 + tid;
  bool act = q < 449;
  __shared__ float Kl[64][64];
  __shared__ float Vl[64][64];
  float qr[64], acc[64];
  size_t qoff = (size_t)(b * 449 + (act ? q : 0)) * 512 + hh * 64;
#pragma unroll
  for (int d = 0; d < 64; ++d) { qr[d] = Qp[qoff + d]; acc[d] = 0.f; }
  float m = -3.0e38f, l = 0.f;

  for (int c0 = 0; c0 < 449; c0 += 64) {
    __syncthreads();
#pragma unroll
    for (int it = 0; it < 16; ++it) {
      int idx = tid + it * 256;
      int kr = idx >> 6, d = idx & 63;
      int kg = c0 + kr;
      size_t off = (size_t)(b * 449 + (kg < 449 ? kg : 0)) * 512 + hh * 64 + d;
      Kl[kr][d] = Kp[off];
      Vl[kr][d] = Vp[off];
    }
    __syncthreads();
    if (act) {
      int jmax = min(64, 449 - c0);
      for (int j = 0; j < jmax; ++j) {
        float s = 0.f;
#pragma unroll
        for (int d = 0; d < 64; ++d) s += qr[d] * Kl[j][d];
        s *= 0.125f;                        // 1/sqrt(64)
        if (s > m) {
          float corr = __expf(m - s);
          l *= corr;
#pragma unroll
          for (int d = 0; d < 64; ++d) acc[d] *= corr;
          m = s;
        }
        float p = __expf(s - m);
        l += p;
#pragma unroll
        for (int d = 0; d < 64; ++d) acc[d] += p * Vl[j][d];
      }
    }
  }
  if (act) {
    float rl = 1.f / l;
    size_t ooff = (size_t)(b * 449 + q) * 512 + hh * 64;
#pragma unroll
    for (int d = 0; d < 64; d += 2) {
      unsigned int pk = (unsigned int)f2bf(acc[d] * rl) |
                        ((unsigned int)f2bf(acc[d + 1] * rl) << 16);
      *reinterpret_cast<unsigned int*>(Obf + ooff + d) = pk;
    }
  }
}

// ---------------------------------------------------------------------------
// Pooling attention: PH=4, pdk=128, 1 query, 448 keys; bf16 out. block 512.
// ---------------------------------------------------------------------------
__global__ void k_pool_attn(const float* __restrict__ qv, const float* __restrict__ Kp,
                            const float* __restrict__ Vp, unsigned short* __restrict__ POb)
{
  int b = blockIdx.x, tid = threadIdx.x;
  __shared__ float sm[4 * 448];
  for (int idx = tid; idx < 1792; idx += 512) {
    int ph = idx / 448, j = idx % 448;
    const float* kr = Kp + (size_t)(b * 448 + j) * 512 + ph * 128;
    const float* qr = qv + ph * 128;
    float s = 0.f;
    for (int d = 0; d < 128; ++d) s += qr[d] * kr[d];
    sm[idx] = s * 0.08838834764831845f;     // 1/sqrt(128)
  }
  __syncthreads();
  int ph = tid >> 7, d = tid & 127;
  float mx = -3.0e38f;
  for (int j = 0; j < 448; ++j) mx = fmaxf(mx, sm[ph * 448 + j]);
  float l = 0.f, a = 0.f;
  for (int j = 0; j < 448; ++j) {
    float p = __expf(sm[ph * 448 + j] - mx);
    l += p;
    a += p * Vp[(size_t)(b * 448 + j) * 512 + ph * 128 + d];
  }
  POb[(size_t)b * 512 + ph * 128 + d] = f2bf(a / l);
}

// ---------------------------------------------------------------------------
// comb[b] = [ h[b,0,:], LN(po[b], pln) ]  (fp32 + bf16)
// ---------------------------------------------------------------------------
__global__ void k_comb(const float* __restrict__ Hb, const float* __restrict__ po,
                       const float* __restrict__ w, const float* __restrict__ bb,
                       float* __restrict__ comb, unsigned short* __restrict__ combb)
{
  int b = blockIdx.x, tid = threadIdx.x;
  for (int c = tid; c < 512; c += 256) {
    float v = Hb[(size_t)b * 449 * 512 + c];
    comb [(size_t)b * 1024 + c] = v;
    combb[(size_t)b * 1024 + c] = f2bf(v);
  }
  __shared__ float r1[256], r2[256];
  float v0 = po[(size_t)b * 512 + tid], v1 = po[(size_t)b * 512 + tid + 256];
  float mean, rinv;
  red2(r1, r2, tid, v0 + v1, v0 * v0 + v1 * v1, 1.f / 512.f, mean, rinv);
  float o0 = (v0 - mean) * rinv * w[tid]       + bb[tid];
  float o1 = (v1 - mean) * rinv * w[tid + 256] + bb[tid + 256];
  comb [(size_t)b * 1024 + 512 + tid]       = o0;
  combb[(size_t)b * 1024 + 512 + tid]       = f2bf(o0);
  comb [(size_t)b * 1024 + 512 + tid + 256] = o1;
  combb[(size_t)b * 1024 + 512 + tid + 256] = f2bf(o1);
}

// ---------------------------------------------------------------------------
// Host orchestration
// ---------------------------------------------------------------------------
static inline int imin(int a, int b) { return a < b ? a : b; }

extern "C" void kernel_launch(void* const* d_in, const int* in_sizes, int n_in,
                              void* d_out, int out_size, void* d_ws, size_t ws_size,
                              hipStream_t stream)
{
  (void)in_sizes; (void)n_in; (void)out_size; (void)ws_size;
  const float* x      = (const float*)d_in[0];
  const float* in_w1  = (const float*)d_in[1];
  const float* in_b1  = (const float*)d_in[2];
  const float* in_w2  = (const float*)d_in[3];
  const float* in_b2  = (const float*)d_in[4];
  const float* in_ln_w= (const float*)d_in[5];
  const float* in_ln_b= (const float*)d_in[6];
  const float* pos    = (const float*)d_in[7];
  const float* cls    = (const float*)d_in[8];
  const float* wq     = (const float*)d_in[9];
  const float* bq     = (const float*)d_in[10];
  const float* wk     = (const float*)d_in[11];
  const float* bk     = (const float*)d_in[12];
  const float* wv     = (const float*)d_in[13];
  const float* bv     = (const float*)d_in[14];
  const float* wo     = (const float*)d_in[15];
  const float* bo     = (const float*)d_in[16];
  const float* conv_w = (const float*)d_in[17];
  const float* conv_b = (const float*)d_in[18];
  const float* aln_w  = (const float*)d_in[19];
  const float* aln_b  = (const float*)d_in[20];
  const float* f_w1   = (const float*)d_in[21];
  const float* f_b1   = (const float*)d_in[22];
  const float* f_w2   = (const float*)d_in[23];
  const float* f_b2   = (const float*)d_in[24];
  const float* f_w3   = (const float*)d_in[25];
  const float* f_b3   = (const float*)d_in[26];
  const float* g_w    = (const float*)d_in[27];
  const float* g_b    = (const float*)d_in[28];
  const float* n1_w   = (const float*)d_in[29];
  const float* n1_b   = (const float*)d_in[30];
  const float* n2_w   = (const float*)d_in[31];
  const float* n2_b   = (const float*)d_in[32];
  const float* pq     = (const float*)d_in[33];
  const float* pqw    = (const float*)d_in[34];
  const float* pqb    = (const float*)d_in[35];
  const float* pkw    = (const float*)d_in[36];
  const float* pkb    = (const float*)d_in[37];
  const float* pvw    = (const float*)d_in[38];
  const float* pvb    = (const float*)d_in[39];
  const float* pow_w  = (const float*)d_in[40];
  const float* pow_b  = (const float*)d_in[41];
  const float* pln_w  = (const float*)d_in[42];
  const float* pln_b  = (const float*)d_in[43];
  const float* h_ln_w = (const float*)d_in[44];
  const float* h_ln_b = (const float*)d_in[45];
  const float* h_w1   = (const float*)d_in[46];
  const float* h_b1   = (const float*)d_in[47];
  const float* h_w2   = (const float*)d_in[48];
  const float* h_b2   = (const float*)d_in[49];
  const float* h_w3   = (const float*)d_in[50];
  const float* h_b3   = (const float*)d_in[51];
  const float* h_w4   = (const float*)d_in[52];
  const float* h_b4   = (const float*)d_in[53];
  const float* u_w1   = (const float*)d_in[54];
  const float* u_b1   = (const float*)d_in[55];
  const float* u_w2   = (const float*)d_in[56];
  const float* u_b2   = (const float*)d_in[57];

  float* out = (float*)d_out;

  const int Bb = 64, Sq = 448, D = 512, Hh = 8, Ll = 6, Tt = 3, Cc = 3,
            FF = 3072, S1 = 449;
  const int M0 = Bb * Sq;   // 28672
  const int M1 = Bb * S1;   // 28736
  const int MC = 3584;      // FFN row chunk
  const size_t PAD = 131072; // bf16 tile-overshoot pad (>=128 rows * 1024)

  char* base = (char*)d_ws;
  size_t off = 0;
  auto allocf = [&](size_t n) { float* p = (float*)(base + off);
                                off += ((n * 4) + 255) & ~(size_t)255; return p; };
  auto allocu = [&](size_t n) { unsigned short* p = (unsigned short*)(base + off);
                                off += ((n * 2) + 255) & ~(size_t)255; return p; };

  // fp32 activations
  float* Hb  = allocf((size_t)M1 * D);
  float* H2  = allocf((size_t)M1 * D);
  float* Qb  = allocf((size_t)M1 * D);
  float* Kb  = allocf((size_t)M1 * D);
  float* Vb  = allocf((size_t)M1 * D);
  float* PQV = allocf(D);
  float* PO2 = allocf((size_t)Bb * D);
  float* COMB= allocf((size_t)Bb * 2 * D);
  float* SBf = allocf((size_t)Bb * 2 * D);   // fp32 head scratch (SB3 / u-mid)

  // bf16 activation shadows (padded for GEMM tile overshoot)
  unsigned short* x_bf   = allocu((size_t)M0 * 128 + PAD);
  unsigned short* T1_bf  = allocu((size_t)M0 * 256 + PAD);
  unsigned short* Hb_bf  = allocu((size_t)M1 * D + PAD);
  unsigned short* H2_bf  = allocu((size_t)M1 * D + PAD);
  unsigned short* Eb_bf  = allocu((size_t)M1 * D + PAD);
  unsigned short* F1_bf  = allocu((size_t)MC * FF + PAD);
  unsigned short* F2_bf  = allocu((size_t)MC * FF + PAD);
  unsigned short* PO_bf  = allocu((size_t)Bb * D + PAD);
  unsigned short* CB_bf  = allocu((size_t)Bb * 2 * D + PAD);
  unsigned short* SB0_bf = allocu((size_t)Bb * 2 * D + PAD);
  unsigned short* SB1_bf = allocu((size_t)Bb * 2 * D + PAD);
  unsigned short* SB2_bf = allocu((size_t)Bb * 2 * D + PAD);

  // bf16 transposed weights
  unsigned short* in_w1t = allocu(128 * 256);
  unsigned short* in_w2t = allocu(256 * 512);
  unsigned short* wqt    = allocu((size_t)Ll * D * D);
  unsigned short* wkt    = allocu((size_t)Ll * D * D);
  unsigned short* wvt    = allocu((size_t)Ll * D * D);
  unsigned short* wot    = allocu((size_t)Ll * D * D);
  unsigned short* gwt    = allocu((size_t)Ll * D * D);
  unsigned short* fw1t   = allocu((size_t)Ll * D * FF);
  unsigned short* fw2t   = allocu((size_t)Ll * FF * FF);
  unsigned short* fw3t   = allocu((size_t)Ll * FF * D);
  unsigned short* pkwt   = allocu((size_t)D * D);
  unsigned short* pvwt   = allocu((size_t)D * D);
  unsigned short* powt   = allocu((size_t)D * D);
  unsigned short* hw1t   = allocu((size_t)Tt * 1024 * 1024);
  unsigned short* hw2t   = allocu((size_t)Tt * 1024 * 512);
  unsigned short* hw3t   = allocu((size_t)Tt * 512 * 256);
  unsigned short* uw1t   = allocu((size_t)Tt * 1024 * 256);

  auto conv_t = [&](const float* src, unsigned short* dst, int K, int N, int nmat) {
    dim3 g((unsigned)(((size_t)K * N + 255) / 256), (unsigned)nmat, 1);
    k_w2bf_t<<<g, 256, 0, stream>>>(src, dst, K, N);
  };
  // ---- one-time weight convert+transpose (amortized over 225 M-tiles) ----
  conv_t(in_w1, in_w1t, 128, 256, 1);
  conv_t(in_w2, in_w2t, 256, 512, 1);
  conv_t(wq, wqt, D, D, Ll);   conv_t(wk, wkt, D, D, Ll);
  conv_t(wv, wvt, D, D, Ll);   conv_t(wo, wot, D, D, Ll);
  conv_t(g_w, gwt, D, D, Ll);
  conv_t(f_w1, fw1t, D, FF, Ll);
  conv_t(f_w2, fw2t, FF, FF, Ll);
  conv_t(f_w3, fw3t, FF, D, Ll);
  conv_t(pkw, pkwt, D, D, 1);  conv_t(pvw, pvwt, D, D, 1);
  conv_t(pow_w, powt, D, D, 1);
  conv_t(h_w1, hw1t, 1024, 1024, Tt);
  conv_t(h_w2, hw2t, 1024, 512, Tt);
  conv_t(h_w3, hw3t, 512, 256, Tt);
  conv_t(u_w1, uw1t, 1024, 256, Tt);
  {
    long n = (long)M0 * 128;
    k_f2bf<<<(unsigned)((n + 255) / 256), 256, 0, stream>>>(x, x_bf, n);
  }

  auto gemm = [&](const unsigned short* A, const unsigned short* Wt, const float* bias,
                  float* Cf, unsigned short* Cbf, int M, int N, int K, int epi,
                  int rowmap) {
    dim3 g(N / 128, (M + 127) / 128, 1);
    k_gemm<<<g, dim3(256), 0, stream>>>(A, Wt, bias, Cf, Cbf, M, N, K, epi, rowmap);
  };

  // ---- input projection ----
  gemm(x_bf,  in_w1t, in_b1, nullptr, T1_bf, M0, 256, 128, EPI_GELU, 0);
  gemm(T1_bf, in_w2t, in_b2, Kb, nullptr, M0, 512, 256, EPI_NONE, 0);
  k_build_h<<<M1, 256, 0, stream>>>(Kb, cls, pos, in_ln_w, in_ln_b, Hb, Hb_bf);

  // ---- transformer layers ----
  for (int i = 0; i < Ll; ++i) {
    const unsigned short* wqt_l = wqt + (size_t)i * D * D;
    const unsigned short* wkt_l = wkt + (size_t)i * D * D;
    const unsigned short* wvt_l = wvt + (size_t)i * D * D;
    const unsigned short* wot_l = wot + (size_t)i * D * D;
    const unsigned short* gwt_l = gwt + (size_t)i * D * D;
    const unsigned short* fw1t_l = fw1t + (size_t)i * D * FF;
    const unsigned short* fw2t_l = fw2t + (size_t)i * FF * FF;
    const unsigned short* fw3t_l = fw3t + (size_t)i * FF * D;
    const float* bq_l = bq + (size_t)i * D;  const float* bk_l = bk + (size_t)i * D;
    const float* bv_l = bv + (size_t)i * D;  const float* bo_l = bo + (size_t)i * D;
    const float* gb_l = g_b + (size_t)i * D;
    const float* fb1_l = f_b1 + (size_t)i * FF;
    const float* fb2_l = f_b2 + (size_t)i * FF;
    const float* fb3_l = f_b3 + (size_t)i * D;
    const float* cw_l = conv_w + (size_t)i * D * 3;
    const float* cb_l = conv_b + (size_t)i * D;
    const float* aw_l = aln_w + (size_t)i * D;  const float* ab_l = aln_b + (size_t)i * D;
    const float* n1w_l = n1_w + (size_t)i * D;  const float* n1b_l = n1_b + (size_t)i * D;
    const float* n2w_l = n2_w + (size_t)i * D;  const float* n2b_l = n2_b + (size_t)i * D;

    gemm(Hb_bf, wqt_l, bq_l, Qb, nullptr, M1, D, D, EPI_NONE, 0);
    gemm(Hb_bf, wkt_l, bk_l, Kb, nullptr, M1, D, D, EPI_NONE, 0);
    gemm(Hb_bf, wvt_l, bv_l, Vb, nullptr, M1, D, D, EPI_NONE, 0);
    k_attn<<<dim3(2, Hh, Bb), 256, 0, stream>>>(Qb, Kb, Vb, Eb_bf);
    gemm(Eb_bf, wot_l, bo_l, Qb, nullptr, M1, D, D, EPI_NONE, 0);  // ga -> Qb
    k_attn_mid<<<M1, 256, 0, stream>>>(Hb, Qb, cw_l, cb_l, aw_l, ab_l,
                                       n1w_l, n1b_l, H2, H2_bf);
    gemm(H2_bf, gwt_l, gb_l, Kb, nullptr, M1, D, D, EPI_SIG, 0);   // gate -> Kb
    for (int r0 = 0; r0 < M1; r0 += MC) {
      int rows = imin(MC, M1 - r0);
      gemm(H2_bf + (size_t)r0 * D, fw1t_l, fb1_l, nullptr, F1_bf, rows, FF, D,  EPI_GELU, 0);
      gemm(F1_bf,                  fw2t_l, fb2_l, nullptr, F2_bf, rows, FF, FF, EPI_GELU, 0);
      gemm(F2_bf,                  fw3t_l, fb3_l, Vb + (size_t)r0 * D, nullptr,
           rows, D, FF, EPI_NONE, 0);
    }
    k_ffn_end<<<M1, 256, 0, stream>>>(H2, Vb, Kb, n2w_l, n2b_l, Hb, Hb_bf);
  }

  // ---- pooling attention ----
  k_small<<<2, 256, 0, stream>>>(pq, pqw, pqb, PQV, 1, 512, 512, 512, EPI_NONE);
  gemm(Hb_bf, pkwt, pkb, Qb, nullptr, M0, D, D, EPI_NONE, 1);   // K_pool (skip-cls)
  gemm(Hb_bf, pvwt, pvb, Kb, nullptr, M0, D, D, EPI_NONE, 1);   // V_pool
  k_pool_attn<<<Bb, 512, 0, stream>>>(PQV, Qb, Kb, PO_bf);
  gemm(PO_bf, powt, pow_b, PO2, nullptr, Bb, D, D, EPI_NONE, 0);
  k_comb<<<Bb, 256, 0, stream>>>(Hb, PO2, pln_w, pln_b, COMB, CB_bf);

  // ---- heads ----
  for (int t = 0; t < Tt; ++t) {
    k_layernorm<<<Bb, 256, 0, stream>>>(COMB, nullptr, SB0_bf,
                                        h_ln_w + (size_t)t * 1024,
                                        h_ln_b + (size_t)t * 1024, 1024);
    gemm(SB0_bf, hw1t + (size_t)t * 1024 * 1024, h_b1 + (size_t)t * 1024,
         nullptr, SB1_bf, Bb, 1024, 1024, EPI_GELU, 0);
    gemm(SB1_bf, hw2t + (size_t)t * 1024 * 512,  h_b2 + (size_t)t * 512,
         nullptr, SB2_bf, Bb, 512, 1024, EPI_GELU, 0);
    gemm(SB2_bf, hw3t + (size_t)t * 512 * 256,   h_b3 + (size_t)t * 256,
         SBf, nullptr, Bb, 256, 512, EPI_GELU, 0);
    k_small<<<(Bb * Cc + 255) / 256, 256, 0, stream>>>(
        SBf, h_w4 + (size_t)t * 256 * 3, h_b4 + (size_t)t * 3,
        out + t * Cc, Bb, Cc, 256, Tt * Cc, EPI_NONE);
    gemm(CB_bf, uw1t + (size_t)t * 1024 * 256, u_b1 + (size_t)t * 256,
         SBf, nullptr, Bb, 256, 1024, EPI_RELU, 0);
    k_small<<<(Bb + 255) / 256, 256, 0, stream>>>(
        SBf, u_w2 + (size_t)t * 256, u_b2 + t,
        out + Bb * Tt * Cc + t, Bb, 1, 256, Tt, EPI_SOFTPLUS);
  }
}
